// MultiHeadSlidingAttention_64218351009955
// MI455X (gfx1250) — matrix operand order repaired and checked by reading, not compiled, not run on hardware
//
#include <hip/hip_runtime.h>
#include <hip/hip_bf16.h>

// MI455X / gfx1250 sliding-window MHA, f16 WMMA (v_wmma_f32_16x16x32_f16).
// All GEMM operands pre-converted to f16 once so the hot loops are pure
// global_load_b128 -> v_wmma (no per-iteration cvt VALU, no WAR hazard NOPs).

typedef __attribute__((ext_vector_type(16))) _Float16 v16h;
typedef __attribute__((ext_vector_type(8)))  _Float16 v8h;
typedef __attribute__((ext_vector_type(8)))  float    v8f;
typedef __attribute__((ext_vector_type(4)))  float    f32x4;

#define DIMC   1024
#define NHEADS 16
#define HDIM   64
#define SEQ    2048
#define BATCH  2
#define WIN    256
#define QTILES (SEQ / 16)   // 128

static __device__ __forceinline__ v8f wmma_f16(v16h a, v16h b, v8f c) {
  return __builtin_amdgcn_wmma_f32_16x16x32_f16(false, a, false, b, (short)0, c,
                                                false, false);
}

// A-fragment assembly: halves 0-7 from `lo`, halves 8-15 from `hi` (each 8
// contiguous f16 = one b128 load).
static __device__ __forceinline__ v16h ld_a_f16(const _Float16* lo,
                                                const _Float16* hi) {
  v8h a = *(const v8h*)lo;
  v8h b = *(const v8h*)hi;
  v16h r;
#pragma unroll
  for (int i = 0; i < 8; ++i) { r[i] = a[i]; r[8 + i] = b[i]; }
  return r;
}

static __device__ __forceinline__ float finite_or_zero(float v) {
  return (v == v && fabsf(v) <= 3.0e38f) ? v : 0.0f;
}

// ---------------------------------------------------------------------------
// Kernel 0: one-pass f32 -> f16 conversion (x and the four weight matrices).
// ---------------------------------------------------------------------------
__global__ void __launch_bounds__(256) cvt_f16_kernel(
    const float* __restrict__ src, _Float16* __restrict__ dst, int n) {
  int i = (blockIdx.x * 256 + threadIdx.x) * 8;
  if (i >= n) return;
  f32x4 a = *(const f32x4*)(src + i);
  f32x4 b = *(const f32x4*)(src + i + 4);
  v8h o;
#pragma unroll
  for (int j = 0; j < 4; ++j) {
    o[j]     = (_Float16)a[j];
    o[4 + j] = (_Float16)b[j];
  }
  *(v8h*)(dst + i) = o;
}

// ---------------------------------------------------------------------------
// Kernel 1: fused QKV projection (+bias, +RMSNorm on Q/K, Q pre-scaled by
// dh^-0.5). One wave computes a 16-token x 64-feature (= one head) tile for
// one of {Q,K,V}.  D(M=feature, N=token) = W x X^T over K=1024.
// ---------------------------------------------------------------------------
__global__ void __launch_bounds__(256) proj_qkv_kernel(
    const _Float16* __restrict__ x16,
    const _Float16* __restrict__ wq16, const float* __restrict__ bq,
    const _Float16* __restrict__ wk16, const float* __restrict__ bk,
    const _Float16* __restrict__ wv16, const float* __restrict__ bv,
    const float* __restrict__ qn, const float* __restrict__ kn,
    _Float16* __restrict__ qbuf, _Float16* __restrict__ kbuf,
    _Float16* __restrict__ vtbuf) {
  const int lane = threadIdx.x & 31;
  int wid = blockIdx.x * 8 + (threadIdx.x >> 5);
  const int which = wid % 3; wid /= 3;          // 0=Q 1=K 2=V
  const int h  = wid % NHEADS; wid /= NHEADS;
  const int lt = wid % QTILES;
  const int b  = wid / QTILES;

  const _Float16* W  = (which == 0) ? wq16 : (which == 1) ? wk16 : wv16;
  const float*    bi = (which == 0) ? bq : (which == 1) ? bk : bv;

  const int token = lt * 16 + (lane & 15);
  // B-fragment (X^T): lane n<16 holds K=k..k+15, lane n>=16 holds k+16..k+31
  const _Float16* xrow =
      x16 + (size_t)(b * SEQ + token) * DIMC + ((lane < 16) ? 0 : 16);
  const int feat  = h * HDIM + (lane & 15);     // W row for m-tile base
  const int alo   = (lane < 16) ? 0 : 16;       // A-frag K segments
  const int ahi   = (lane < 16) ? 8 : 24;
  const int rbase = (lane >= 16) ? 8 : 0;

  v8f acc[4] = {v8f{}, v8f{}, v8f{}, v8f{}};
#pragma unroll 2
  for (int k = 0; k < DIMC; k += 32) {
    __builtin_prefetch(xrow + k + 256, 0, 0);   // global_prefetch_b8
    v16h bf = *(const v16h*)(xrow + k);
    const _Float16* w0 = W + (size_t)feat * DIMC + k;
#pragma unroll
    for (int t = 0; t < 4; ++t) {
      const _Float16* wr = w0 + (size_t)t * 16 * DIMC;
      acc[t] = wmma_f16(ld_a_f16(wr + alo, wr + ahi), bf, acc[t]);
    }
  }

  // Epilogue: bias (+norm for Q/K), store.
  float vals[4][8];
#pragma unroll
  for (int t = 0; t < 4; ++t) {
    const float* bp = bi + h * HDIM + t * 16 + rbase;
    f32x4 b0 = *(const f32x4*)bp, b1 = *(const f32x4*)(bp + 4);
#pragma unroll
    for (int r = 0; r < 8; ++r)
      vals[t][r] = acc[t][r] + ((r < 4) ? b0[r] : b1[r - 4]);
  }

  if (which < 2) {
    const float* wn = (which == 0) ? qn : kn;
    float ss = 0.0f;
#pragma unroll
    for (int t = 0; t < 4; ++t)
#pragma unroll
      for (int r = 0; r < 8; ++r) {
        float c = fminf(fmaxf(vals[t][r], -10000.0f), 10000.0f);
        vals[t][r] = c;
        ss += c * c;
      }
    ss += __shfl_xor(ss, 16, 32);               // partner lane has feats +8
    float rms = sqrtf(fmaxf(ss * (1.0f / HDIM), 1e-5f) + 1e-5f);
    float inv = 1.0f / rms;
    if (which == 0) inv *= 0.125f;              // fold dh^-0.5 into Q
    _Float16* ob =
        ((which == 0) ? qbuf : kbuf) +
        ((size_t)(b * NHEADS + h) * SEQ + token) * HDIM;
#pragma unroll
    for (int t = 0; t < 4; ++t) {
      const float* wp = wn + t * 16 + rbase;
      f32x4 w0 = *(const f32x4*)wp, w1 = *(const f32x4*)(wp + 4);
      v8h o;
#pragma unroll
      for (int r = 0; r < 8; ++r) {
        float v = vals[t][r] * inv * ((r < 4) ? w0[r] : w1[r - 4]);
        o[r] = (_Float16)finite_or_zero(v);
      }
      *(v8h*)(ob + t * 16 + rbase) = o;
    }
  } else {
    // V stored transposed: vt[(b,h,dh,l)] so it becomes an A-fragment later.
#pragma unroll
    for (int t = 0; t < 4; ++t)
#pragma unroll
      for (int r = 0; r < 8; ++r) {
        int dh = t * 16 + rbase + r;
        vtbuf[((size_t)(b * NHEADS + h) * HDIM + dh) * SEQ + token] =
            (_Float16)vals[t][r];
      }
  }
}

// ---------------------------------------------------------------------------
// Kernel 2: flash-style sliding-window attention.  One wave per
// (b, h, 16-query tile).  S^T(key,query) = K x Q^T ; O^T += V^T x P^T.
// ---------------------------------------------------------------------------
__global__ void __launch_bounds__(256) attn_swin_kernel(
    const _Float16* __restrict__ qbuf, const _Float16* __restrict__ kbuf,
    const _Float16* __restrict__ vtbuf, _Float16* __restrict__ abuf) {
  const int lane = threadIdx.x & 31;
  int wid = blockIdx.x * 8 + (threadIdx.x >> 5);
  const int qt = wid % QTILES; wid /= QTILES;
  const int h  = wid % NHEADS;
  const int b  = wid / NHEADS;
  const int bh = b * NHEADS + h;
  const int q0 = qt * 16;
  const int query = q0 + (lane & 15);

  const int alo   = (lane < 16) ? 0 : 16;
  const int ahi   = (lane < 16) ? 8 : 24;
  const int rbase = (lane >= 16) ? 8 : 0;

  // Q^T B-fragments (k = dh), one per 32-wide dh chunk.
  const _Float16* qrow =
      qbuf + ((size_t)bh * SEQ + query) * HDIM + ((lane < 16) ? 0 : 16);
  v16h qf0 = *(const v16h*)(qrow);
  v16h qf1 = *(const v16h*)(qrow + 32);

  v8f acc[4] = {v8f{}, v8f{}, v8f{}, v8f{}};   // O^T, dh tiles of 16
  float m = -1e30f, lsum = 0.0f;

  int first = q0 - (WIN - 1); if (first < 0) first = 0;
  first &= ~31;                                 // 32-key blocks, mask fixes up
  const int kend = q0 + 16;

  for (int kb = first; kb < kend; kb += 32) {
    v8f S0 = v8f{}, S1 = v8f{};
    {
      int key0 = kb + (lane & 15); if (key0 > SEQ - 1) key0 = SEQ - 1;
      const _Float16* kr0 = kbuf + ((size_t)bh * SEQ + key0) * HDIM;
      S0 = wmma_f16(ld_a_f16(kr0 + alo, kr0 + ahi), qf0, S0);
      S0 = wmma_f16(ld_a_f16(kr0 + 32 + alo, kr0 + 32 + ahi), qf1, S0);
      int key1 = kb + 16 + (lane & 15); if (key1 > SEQ - 1) key1 = SEQ - 1;
      const _Float16* kr1 = kbuf + ((size_t)bh * SEQ + key1) * HDIM;
      S1 = wmma_f16(ld_a_f16(kr1 + alo, kr1 + ahi), qf0, S1);
      S1 = wmma_f16(ld_a_f16(kr1 + 32 + alo, kr1 + 32 + ahi), qf1, S1);
    }

    // Causal + window mask, tile max.
    float p0[8], p1[8];
    float tmax = -1e30f;
#pragma unroll
    for (int r = 0; r < 8; ++r) {
      int k0 = kb + r + rbase;
      int k1 = k0 + 16;
      float s0 = S0[r], s1 = S1[r];
      if (k0 > query || (query - k0) >= WIN) s0 = -1e30f;
      if (k1 > query || (query - k1) >= WIN) s1 = -1e30f;
      p0[r] = s0; p1[r] = s1;
      tmax = fmaxf(tmax, fmaxf(s0, s1));
    }
    tmax = fmaxf(tmax, __shfl_xor(tmax, 16, 32));
    float mnew  = fmaxf(m, tmax);
    float alpha = __expf(m - mnew);
    m = mnew;

    float rs = 0.0f;
#pragma unroll
    for (int r = 0; r < 8; ++r) {
      p0[r] = __expf(p0[r] - mnew);
      p1[r] = __expf(p1[r] - mnew);
      rs += p0[r] + p1[r];
    }
    rs += __shfl_xor(rs, 16, 32);
    lsum = lsum * alpha + rs;
#pragma unroll
    for (int t = 0; t < 4; ++t)
#pragma unroll
      for (int r = 0; r < 8; ++r) acc[t][r] *= alpha;

    // Rebuild P^T as a B-fragment (k = key 0..31, n = query).
    v16h pf;
#pragma unroll
    for (int r = 0; r < 8; ++r) {
      float o0 = __shfl_xor(p0[r], 16, 32);
      float o1 = __shfl_xor(p1[r], 16, 32);
      pf[r]     = (_Float16)((lane < 16) ? p0[r] : o1);
      pf[8 + r] = (_Float16)((lane < 16) ? o0 : p1[r]);
    }

    // O^T += V^T x P^T
    int klo = kb + alo; if (klo > SEQ - 8) klo = SEQ - 8;
    int khi = kb + ahi; if (khi > SEQ - 8) khi = SEQ - 8;
#pragma unroll
    for (int t = 0; t < 4; ++t) {
      const _Float16* vr =
          vtbuf + ((size_t)bh * HDIM + t * 16 + (lane & 15)) * SEQ;
      acc[t] = wmma_f16(ld_a_f16(vr + klo, vr + khi), pf, acc[t]);
    }
  }

  float inv = (lsum > 0.0f) ? (1.0f / lsum) : 0.0f;
  _Float16* ob =
      abuf + ((size_t)(b * SEQ + query)) * DIMC + h * HDIM + rbase;
#pragma unroll
  for (int t = 0; t < 4; ++t) {
    v8h o;
#pragma unroll
    for (int r = 0; r < 8; ++r)
      o[r] = (_Float16)finite_or_zero(acc[t][r] * inv);
    *(v8h*)(ob + t * 16) = o;
  }
}

// ---------------------------------------------------------------------------
// Kernel 3: output projection, f16 activations x f16 Wo, f32 out + bias.
// ---------------------------------------------------------------------------
__global__ void __launch_bounds__(256) oproj_kernel(
    const _Float16* __restrict__ abuf, const _Float16* __restrict__ wo16,
    const float* __restrict__ bo, float* __restrict__ out) {
  const int lane = threadIdx.x & 31;
  int wid = blockIdx.x * 8 + (threadIdx.x >> 5);
  const int fb = wid % (DIMC / 64); wid /= (DIMC / 64);
  const int lt = wid % QTILES;
  const int b  = wid / QTILES;
  const int token = lt * 16 + (lane & 15);
  const int feat  = fb * 64 + (lane & 15);
  const _Float16* arow =
      abuf + (size_t)(b * SEQ + token) * DIMC + ((lane < 16) ? 0 : 16);
  const int alo = (lane < 16) ? 0 : 16;
  const int ahi = (lane < 16) ? 8 : 24;

  v8f acc[4] = {v8f{}, v8f{}, v8f{}, v8f{}};
#pragma unroll 2
  for (int k = 0; k < DIMC; k += 32) {
    __builtin_prefetch(arow + k + 256, 0, 0);
    v16h bf = *(const v16h*)(arow + k);
    const _Float16* w0 = wo16 + (size_t)feat * DIMC + k;
#pragma unroll
    for (int t = 0; t < 4; ++t) {
      const _Float16* wr = w0 + (size_t)t * 16 * DIMC;
      acc[t] = wmma_f16(ld_a_f16(wr + alo, wr + ahi), bf, acc[t]);
    }
  }

  const int rbase = (lane >= 16) ? 8 : 0;
  float* orow = out + (size_t)(b * SEQ + token) * DIMC + fb * 64 + rbase;
#pragma unroll
  for (int t = 0; t < 4; ++t) {
    const float* bp = bo + fb * 64 + t * 16 + rbase;
    f32x4 b0 = *(const f32x4*)bp, b1 = *(const f32x4*)(bp + 4);
    f32x4 o0, o1;
#pragma unroll
    for (int r = 0; r < 4; ++r) {
      o0[r] = acc[t][r] + b0[r];
      o1[r] = acc[t][4 + r] + b1[r];
    }
    *(f32x4*)(orow + t * 16)     = o0;
    *(f32x4*)(orow + t * 16 + 4) = o1;
  }
}

extern "C" void kernel_launch(void* const* d_in, const int* in_sizes, int n_in,
                              void* d_out, int out_size, void* d_ws,
                              size_t ws_size, hipStream_t stream) {
  (void)in_sizes; (void)n_in; (void)out_size; (void)ws_size;
  const float* x  = (const float*)d_in[0];
  const float* wq = (const float*)d_in[1];
  const float* bq = (const float*)d_in[2];
  const float* wk = (const float*)d_in[3];
  const float* bk = (const float*)d_in[4];
  const float* wv = (const float*)d_in[5];
  const float* bv = (const float*)d_in[6];
  const float* wo = (const float*)d_in[7];
  const float* bo = (const float*)d_in[8];
  const float* qn = (const float*)d_in[9];
  const float* kn = (const float*)d_in[10];
  float* out = (float*)d_out;

  // Workspace partition (f16 buffers):
  //   qbuf/kbuf/vtbuf/abuf: 8 MiB each; x16: 8 MiB; wq/wk/wv/wo16: 2 MiB each.
  char* ws = (char*)d_ws;
  const size_t seg  = (size_t)BATCH * NHEADS * SEQ * HDIM * sizeof(_Float16);
  const size_t wseg = (size_t)DIMC * DIMC * sizeof(_Float16);
  _Float16* qbuf  = (_Float16*)(ws);
  _Float16* kbuf  = (_Float16*)(ws + seg);
  _Float16* vtbuf = (_Float16*)(ws + 2 * seg);
  _Float16* abuf  = (_Float16*)(ws + 3 * seg);
  _Float16* x16   = (_Float16*)(ws + 4 * seg);
  _Float16* wq16  = (_Float16*)(ws + 5 * seg);
  _Float16* wk16  = (_Float16*)(ws + 5 * seg + wseg);
  _Float16* wv16  = (_Float16*)(ws + 5 * seg + 2 * wseg);
  _Float16* wo16  = (_Float16*)(ws + 5 * seg + 3 * wseg);

  const int nw = DIMC * DIMC;                 // 1M elems -> 512 blocks
  const int nx = BATCH * SEQ * DIMC;          // 4M elems -> 2048 blocks
  cvt_f16_kernel<<<nx / (8 * 256), 256, 0, stream>>>(x, x16, nx);
  cvt_f16_kernel<<<nw / (8 * 256), 256, 0, stream>>>(wq, wq16, nw);
  cvt_f16_kernel<<<nw / (8 * 256), 256, 0, stream>>>(wk, wk16, nw);
  cvt_f16_kernel<<<nw / (8 * 256), 256, 0, stream>>>(wv, wv16, nw);
  cvt_f16_kernel<<<nw / (8 * 256), 256, 0, stream>>>(wo, wo16, nw);

  // 12288 wave-tasks, 8 waves per 256-thread block.
  proj_qkv_kernel<<<(BATCH * QTILES * NHEADS * 3) / 8, 256, 0, stream>>>(
      x16, wq16, bq, wk16, bk, wv16, bv, qn, kn, qbuf, kbuf, vtbuf);
  // 4096 wave-tasks.
  attn_swin_kernel<<<(BATCH * NHEADS * QTILES) / 8, 256, 0, stream>>>(
      qbuf, kbuf, vtbuf, abuf);
  // 4096 wave-tasks.
  oproj_kernel<<<(BATCH * QTILES * (DIMC / 64)) / 8, 256, 0, stream>>>(
      abuf, wo16, bo, out);
}
